// UNetDecoder_with_features_32074815767209
// MI455X (gfx1250) — matrix operand order repaired and checked by reading, not compiled, not run on hardware
//
#include <hip/hip_runtime.h>
#include <hip/hip_bf16.h>

// ---------------------------------------------------------------------------
// ViG Grapher+FFN block for MI455X (gfx1250, wave32, WMMA bf16 16x16x32).
// Sizes fixed per reference: B=2, C=128, N=32768, K=9, H=512, BN=65536.
// GEMM B-operand staging uses GLOBAL_LOAD_ASYNC_TO_LDS_B128 (ASYNCcnt),
// double-buffered in LDS.
// ---------------------------------------------------------------------------

typedef __attribute__((ext_vector_type(16))) __bf16 v16bf;
typedef __attribute__((ext_vector_type(8)))  float  v8f;
typedef __attribute__((ext_vector_type(4)))  float  v4f;
typedef __attribute__((ext_vector_type(4)))  unsigned int v4u;

#define BB   2
#define CC   128
#define NN   32768
#define KK   9
#define HH   512
#define BN   65536   // B*N columns
#define EPSV 1e-5f

__device__ __forceinline__ unsigned short f2bf(float f) {
  unsigned int u = __float_as_uint(f);
  u += 0x7FFFu + ((u >> 16) & 1u);           // round-to-nearest-even
  return (unsigned short)(u >> 16);
}

// async global->LDS 16B copy (GVS mode: saddr base + 32-bit vgpr byte offset).
// lds_off is the LDS byte offset within the workgroup allocation (low 32 bits
// of the flat shared address, per the gfx1250 LDS aperture rule).
__device__ __forceinline__ void async_copy_b128(unsigned lds_off,
                                                const void* sbase,
                                                int voff_bytes) {
  asm volatile("global_load_async_to_lds_b128 %0, %1, %2 offset:0"
               :: "v"(lds_off), "v"(voff_bytes), "s"(sbase)
               : "memory");
}

// ---------------------------------------------------------------- utilities
__global__ void zero_kernel(float* p, int n) {
  int i = blockIdx.x * blockDim.x + threadIdx.x;
  if (i < n) p[i] = 0.0f;
}

__global__ void convert_bf16_kernel(const float* __restrict__ w,
                                    unsigned short* __restrict__ o, int n) {
  int i = blockIdx.x * blockDim.x + threadIdx.x;
  if (i < n) o[i] = f2bf(w[i]);
}

// x [B,C,N] f32  ->  xT [B,N,C] f32   (LDS 64x64 tile transpose)
__global__ __launch_bounds__(256) void transpose_x_kernel(
    const float* __restrict__ x, float* __restrict__ xT) {
  __shared__ float lds[64 * 65];
  int b  = blockIdx.z;
  int n0 = blockIdx.x * 64;
  int c0 = blockIdx.y * 64;
  int tid = threadIdx.x;
  for (int i = 0; i < 16; ++i) {
    int idx = i * 256 + tid;
    int nl = idx & 63, cl = idx >> 6;           // nl fastest -> coalesced read
    lds[nl * 65 + cl] = x[((size_t)b * CC + c0 + cl) * NN + n0 + nl];
  }
  __syncthreads();
  for (int i = 0; i < 16; ++i) {
    int idx = i * 256 + tid;
    int cl = idx & 63, nl = idx >> 6;           // cl fastest -> coalesced write
    xT[((size_t)b * NN + n0 + nl) * CC + c0 + cl] = lds[nl * 65 + cl];
  }
}

// Gather + max-relative + interleave -> feat bf16 [2C rows, BN cols]
__global__ __launch_bounds__(256) void build_feat_kernel(
    const float* __restrict__ xT,          // [B,N,C]
    const int*   __restrict__ edge,        // [2,B,N,K]
    unsigned short* __restrict__ feat) {   // [2C, BN]
  __shared__ int   i0[64 * KK];
  __shared__ int   i1[64 * KK];
  __shared__ float relbuf[64 * 129];
  __shared__ float xbuf[64 * 129];
  const int blocksPerB = NN / 64;
  int b  = blockIdx.x / blocksPerB;
  int n0 = (blockIdx.x % blocksPerB) * 64;
  int tid = threadIdx.x;

  for (int i = tid; i < 64 * KK; i += 256) {   // contiguous index loads
    i0[i] = edge[(((size_t)0 * BB + b) * NN + n0) * KK + i];
    i1[i] = edge[(((size_t)1 * BB + b) * NN + n0) * KK + i];
  }
  __syncthreads();

  const float* xTb = xT + (size_t)b * NN * CC;
  for (int it = 0; it < 32; ++it) {
    int c  = tid & 127;                         // c fastest: coalesced gathers
    int nl = it * 2 + (tid >> 7);
    float xs  = xTb[(size_t)(n0 + nl) * CC + c];
    float rel = -3.0e38f;
    for (int k = 0; k < KK; ++k) {
      int j = i0[nl * KK + k];                  // neighbor (x_j)
      int i = i1[nl * KK + k];                  // center   (x_i)
      rel = fmaxf(rel, xTb[(size_t)j * CC + c] - xTb[(size_t)i * CC + c]);
    }
    relbuf[nl * 129 + c] = rel;
    xbuf[nl * 129 + c]   = xs;
  }
  __syncthreads();

  size_t colbase = (size_t)b * NN + n0;
  for (int it = 0; it < 64; ++it) {
    int idx = it * 256 + tid;
    int nl = idx & 63;                          // nl fastest -> coalesced write
    int r  = idx >> 6;                          // feat row 0..255
    int c  = r >> 1;
    float v = (r & 1) ? relbuf[nl * 129 + c] : xbuf[nl * 129 + c];
    feat[(size_t)r * BN + colbase + nl] = f2bf(v);
  }
}

// ---------------------------------------------------------------- WMMA GEMM
// D[col, m] = sum_k A[m,k] * Bact[k,col]      (bf16 x bf16 -> f32)
// A : bf16 [M, K] row-major (weights, L2-resident)
// B : bf16 [K, NCOLS] row-major (activations), staged async to LDS
// D : f32  [NCOLS, M]
// Block: 256 thr = 8 waves; WG tile = 128 rows x 128 cols; wave = 16 x 128.
#define LDSB_STRIDE 136                         // 128 cols + pad (272B = 17*16B)
__global__ __launch_bounds__(256) void gemm_bf16_kernel(
    const unsigned short* __restrict__ A,
    const unsigned short* __restrict__ Bact,
    float* __restrict__ D,
    int M, int K, int NCOLS) {
  __shared__ __align__(16) unsigned short ldsB[2][32 * LDSB_STRIDE];
  const int col0  = blockIdx.x * 128;
  const int m0blk = blockIdx.y * 128;
  const int tid  = threadIdx.x;
  const int wave = tid >> 5;
  const int lane = tid & 31;
  const int mrow0 = m0blk + wave * 16;

  union FragA { v16bf v; v4u q[2]; };
  union FragB { v16bf v; v4u q[2]; };
  union Acc   { v8f v; v4f f[2]; float s[8]; };
  Acc acc[8];
  for (int t = 0; t < 8; ++t)
    for (int r = 0; r < 8; ++r) acc[t].s[r] = 0.0f;

  const int arow = mrow0 + (lane & 15);
  const int kb   = (lane >> 4) * 8;

  // staging geometry: thread owns k-row (tid>>3) and 32B col segment
  const int srow = tid >> 3;                    // 0..31
  const int scol = (tid & 7) * 16;              // bf16 column 0,16,..,112
  const unsigned lds0 =
      (unsigned)(unsigned long long)(&ldsB[0][0]);  // LDS byte offset (aperture low bits)

  // prologue: stage chunk 0 into buffer 0
  {
    int goff = (int)((((size_t)srow) * NCOLS + col0 + scol) * 2);
    unsigned l = lds0 + (unsigned)(srow * LDSB_STRIDE + scol) * 2;
    async_copy_b128(l, Bact, goff);
    async_copy_b128(l + 16, Bact, goff + 16);
  }

  const int nk = K >> 5;
  for (int i = 0; i < nk; ++i) {
    if (i + 1 < nk) {
      int kc2 = (i + 1) << 5;
      int buf2 = (i + 1) & 1;
      int goff = (int)((((size_t)(kc2 + srow)) * NCOLS + col0 + scol) * 2);
      unsigned l = lds0 + (unsigned)(buf2 * 32 * LDSB_STRIDE +
                                     srow * LDSB_STRIDE + scol) * 2;
      async_copy_b128(l, Bact, goff);
      async_copy_b128(l + 16, Bact, goff + 16);
      asm volatile("s_wait_asynccnt 0x2" ::: "memory");  // oldest chunk landed
    } else {
      asm volatile("s_wait_asynccnt 0x0" ::: "memory");
    }
    __syncthreads();  // all waves' staging of current buffer visible

    const unsigned short* lb = &ldsB[i & 1][0];
    const int kc = i << 5;

    FragA a;  // ISA 16-bit A layout
    a.q[0] = *(const v4u*)(A + (size_t)arow * K + kc + kb);
    a.q[1] = *(const v4u*)(A + (size_t)arow * K + kc + kb + 16);

    FragB bf[8];
#pragma unroll
    for (int ct = 0; ct < 8; ++ct) {            // preload all B fragments
      for (int s = 0; s < 8; ++s) {
        ((unsigned short*)&bf[ct])[s] = lb[(kb + s) * LDSB_STRIDE + ct * 16 + (lane & 15)];
        ((unsigned short*)&bf[ct])[s + 8] = lb[(kb + 16 + s) * LDSB_STRIDE + ct * 16 + (lane & 15)];
      }
    }
#pragma unroll
    for (int ct = 0; ct < 8; ++ct) {            // back-to-back WMMA burst
      acc[ct].v = __builtin_amdgcn_wmma_f32_16x16x32_bf16(
          false, a.v, false, bf[ct].v, (short)0, acc[ct].v, false, false);
    }
    __syncthreads();  // reads done before this buffer is overwritten
  }

  // D layout [col][m]: vgpr r, lane l -> (m = 8*(l>>4)+r, n = l&15)
  const int ml = (lane >> 4) * 8;
  const int nl = lane & 15;
  for (int ct = 0; ct < 8; ++ct) {
    size_t col = (size_t)col0 + ct * 16 + nl;
    v4f* p = (v4f*)(D + col * M + mrow0 + ml);
    p[0] = acc[ct].f[0];
    p[1] = acc[ct].f[1];
  }
}

// ---------------------------------------------------------------- BN pieces
__global__ __launch_bounds__(256) void stats_kernel(
    const float* __restrict__ pre, float* __restrict__ sum,
    float* __restrict__ sumsq, int M, int cols_per_block) {
  int col0 = blockIdx.x * cols_per_block;
  for (int ch = threadIdx.x; ch < M; ch += 256) {
    float s = 0.0f, q = 0.0f;
    for (int c = 0; c < cols_per_block; ++c) {
      float v = pre[(size_t)(col0 + c) * M + ch];
      s += v; q += v * v;
    }
    atomicAdd(&sum[ch], s);
    atomicAdd(&sumsq[ch], q);
  }
}

__global__ void finalize_kernel(const float* __restrict__ sum,
                                const float* __restrict__ sq,
                                const float* __restrict__ g,
                                const float* __restrict__ be,
                                float* __restrict__ scale,
                                float* __restrict__ shift,
                                int M, float inv_count) {
  int ch = blockIdx.x * blockDim.x + threadIdx.x;
  if (ch < M) {
    float mean = sum[ch] * inv_count;
    float var  = sq[ch] * inv_count - mean * mean;
    float sc   = g[ch] * rsqrtf(var + EPSV);
    scale[ch] = sc;
    shift[ch] = be[ch] - mean * sc;
  }
}

// act[m, col] = bf16( relu(scale[m]*pre[col,m] + shift[m]) )   (transpose)
__global__ __launch_bounds__(256) void apply_bn_relu_T_kernel(
    const float* __restrict__ pre, const float* __restrict__ scale,
    const float* __restrict__ shift, unsigned short* __restrict__ act,
    int M, int ncols) {
  __shared__ float lds[64 * 65];
  int col0 = blockIdx.x * 64;
  int m0   = blockIdx.y * 64;
  int tid  = threadIdx.x;
  for (int i = 0; i < 16; ++i) {
    int idx = i * 256 + tid;
    int ml = idx & 63, cl = idx >> 6;           // ml fastest -> coalesced read
    float v = pre[(size_t)(col0 + cl) * M + m0 + ml];
    v = fmaxf(fmaf(scale[m0 + ml], v, shift[m0 + ml]), 0.0f);
    lds[ml * 65 + cl] = v;
  }
  __syncthreads();
  for (int i = 0; i < 8; ++i) {
    int idx = i * 256 + tid;
    int cp = idx & 31, ml = idx >> 5;           // packed bf16x2 stores
    unsigned int lo = f2bf(lds[ml * 65 + cp * 2]);
    unsigned int hi = f2bf(lds[ml * 65 + cp * 2 + 1]);
    *(unsigned int*)(act + (size_t)(m0 + ml) * ncols + col0 + cp * 2) =
        lo | (hi << 16);
  }
}

// out[b,c,n] = sc3*opre + sh3 + relu(sc1*ypre + sh1)   (transpose to [B,C,N])
__global__ __launch_bounds__(256) void epilogue_kernel(
    const float* __restrict__ ypre, const float* __restrict__ opre,
    const float* __restrict__ sc1, const float* __restrict__ sh1,
    const float* __restrict__ sc3, const float* __restrict__ sh3,
    float* __restrict__ out) {
  __shared__ float lds[64 * 65];
  int col0 = blockIdx.x * 64;                   // over BN (stays within one b)
  int c0   = blockIdx.y * 64;
  int tid  = threadIdx.x;
  for (int i = 0; i < 16; ++i) {
    int idx = i * 256 + tid;
    int cl = idx & 63, nl = idx >> 6;           // cl fastest -> coalesced read
    size_t p = (size_t)(col0 + nl) * CC + c0 + cl;
    float y = fmaxf(fmaf(sc1[c0 + cl], ypre[p], sh1[c0 + cl]), 0.0f);
    float o = fmaf(sc3[c0 + cl], opre[p], sh3[c0 + cl]);
    lds[cl * 65 + nl] = y + o;
  }
  __syncthreads();
  int b  = col0 >> 15;                          // col0 / N
  int n0 = col0 & (NN - 1);
  for (int i = 0; i < 16; ++i) {
    int idx = i * 256 + tid;
    int nl = idx & 63, cl = idx >> 6;           // nl fastest -> coalesced write
    out[((size_t)b * CC + c0 + cl) * NN + n0 + nl] = lds[cl * 65 + nl];
  }
}

// ---------------------------------------------------------------- launcher
extern "C" void kernel_launch(void* const* d_in, const int* in_sizes, int n_in,
                              void* d_out, int out_size, void* d_ws, size_t ws_size,
                              hipStream_t stream) {
  const float* x     = (const float*)d_in[0];
  const int*   edge  = (const int*)  d_in[1];
  const float* w_mr  = (const float*)d_in[2];
  // b_mr/b1/b2 (d_in[3],[7],[11]) cancel exactly under training-mode BN.
  const float* g_mr  = (const float*)d_in[4];
  const float* be_mr = (const float*)d_in[5];
  const float* w1    = (const float*)d_in[6];
  const float* g1    = (const float*)d_in[8];
  const float* be1   = (const float*)d_in[9];
  const float* w2    = (const float*)d_in[10];
  const float* g2    = (const float*)d_in[12];
  const float* be2   = (const float*)d_in[13];
  float* out = (float*)d_out;

  char* ws = (char*)d_ws;
  const size_t MB = 1ull << 20;
  unsigned short* wmr_bf = (unsigned short*)(ws + 0);            //  64 KB
  unsigned short* w1_bf  = (unsigned short*)(ws + 65536);        // 128 KB
  unsigned short* w2_bf  = (unsigned short*)(ws + 65536 + 131072);
  float* stats = (float*)(ws + 327680);                          //  24 KB
  float* sum1 = stats + 0 * 512, * sq1 = stats + 1 * 512;
  float* sum2 = stats + 2 * 512, * sq2 = stats + 3 * 512;
  float* sum3 = stats + 4 * 512, * sq3 = stats + 5 * 512;
  float* sc1  = stats + 6 * 512, * sh1 = stats + 7 * 512;
  float* sc2  = stats + 8 * 512, * sh2 = stats + 9 * 512;
  float* sc3  = stats + 10 * 512, * sh3 = stats + 11 * 512;

  float*          xT   = (float*)(ws + 1 * MB);                  // 32 MB
  float*          opre = xT;                                     // reuse (xT dead)
  unsigned short* feat = (unsigned short*)(ws + 33 * MB);        // 32 MB
  unsigned short* yact = feat;                                   // reuse (feat dead)
  float*          ypre = (float*)(ws + 65 * MB);                 // 32 MB
  float*          hpre = (float*)(ws + 97 * MB);                 // 128 MB
  unsigned short* hact = (unsigned short*)(ws + 225 * MB);       // 64 MB

  const float invBN = 1.0f / (float)BN;

  zero_kernel<<<12, 256, 0, stream>>>(stats, 12 * 512);
  convert_bf16_kernel<<<128, 256, 0, stream>>>(w_mr, wmr_bf, CC * 2 * CC);
  convert_bf16_kernel<<<256, 256, 0, stream>>>(w1, w1_bf, HH * CC);
  convert_bf16_kernel<<<256, 256, 0, stream>>>(w2, w2_bf, CC * HH);

  transpose_x_kernel<<<dim3(NN / 64, CC / 64, BB), 256, 0, stream>>>(x, xT);
  build_feat_kernel<<<BB * (NN / 64), 256, 0, stream>>>(xT, edge, feat);

  // GEMM1: [128 x 65536] = w_mr[128,256] x feat[256,65536]
  gemm_bf16_kernel<<<dim3(BN / 128, 1), 256, 0, stream>>>(wmr_bf, feat, ypre, CC, 2 * CC, BN);
  stats_kernel<<<BN / 256, 256, 0, stream>>>(ypre, sum1, sq1, CC, 256);
  finalize_kernel<<<2, 256, 0, stream>>>(sum1, sq1, g_mr, be_mr, sc1, sh1, CC, invBN);
  apply_bn_relu_T_kernel<<<dim3(BN / 64, CC / 64), 256, 0, stream>>>(ypre, sc1, sh1, yact, CC, BN);

  // GEMM2: [512 x 65536] = w1[512,128] x yact[128,65536]
  gemm_bf16_kernel<<<dim3(BN / 128, HH / 128), 256, 0, stream>>>(w1_bf, yact, hpre, HH, CC, BN);
  stats_kernel<<<BN / 256, 256, 0, stream>>>(hpre, sum2, sq2, HH, 256);
  finalize_kernel<<<2, 256, 0, stream>>>(sum2, sq2, g1, be1, sc2, sh2, HH, invBN);
  apply_bn_relu_T_kernel<<<dim3(BN / 64, HH / 64), 256, 0, stream>>>(hpre, sc2, sh2, hact, HH, BN);

  // GEMM3: [128 x 65536] = w2[128,512] x hact[512,65536]
  gemm_bf16_kernel<<<dim3(BN / 128, 1), 256, 0, stream>>>(w2_bf, hact, opre, CC, HH, BN);
  stats_kernel<<<BN / 256, 256, 0, stream>>>(opre, sum3, sq3, CC, 256);
  finalize_kernel<<<2, 256, 0, stream>>>(sum3, sq3, g2, be2, sc3, sh3, CC, invBN);

  epilogue_kernel<<<dim3(BN / 64, CC / 64), 256, 0, stream>>>(ypre, opre, sc1, sh1, sc3, sh3, out);
}
